// MultiHeadAttention_52261162058207
// MI455X (gfx1250) — matrix operand order repaired and checked by reading, not compiled, not run on hardware
//
#include <hip/hip_runtime.h>
#include <hip/hip_bf16.h>

typedef __bf16 bf16;
typedef __attribute__((ext_vector_type(16))) __bf16 v16bf;
typedef __attribute__((ext_vector_type(8)))  __bf16 v8bf;
typedef __attribute__((ext_vector_type(4)))  __bf16 v4bf;
typedef __attribute__((ext_vector_type(2)))  __bf16 v2bf;
typedef __attribute__((ext_vector_type(8)))  float  v8f;

#define DEV __device__ __forceinline__

// ---- problem constants ----
static constexpr int Bt = 4;        // batch
static constexpr int Sq = 2048;     // sequence
static constexpr int Dm = 1024;     // model dim
static constexpr int Hh = 16;       // heads
static constexpr int Dh = 64;       // head dim
static constexpr int Ee = 1024;     // h*d

// SCALE * log2(e): folded into Q projection so attention logits are in log2 domain
#define QSCALE 0.1803368801111244f

// ---- helpers ----
static DEV bf16 f2bf(float f) { return (bf16)f; }   // native fptrunc -> v_cvt bf16

// A-matrix 16x32 bf16 fragment from row-major [16 x K], K contiguous.
// Lane L: row = L&15; halves 0..7 -> K = g*8.. ; halves 8..15 -> K = 16+g*8.. (g = L>>4)
template <typename P>
static DEV v16bf load_a(P base, int ld, int lane) {
  int r = lane & 15, g = lane >> 4;
  P p = base + (size_t)r * ld + g * 8;
  v8bf lo = *(const v8bf*)(p);
  v8bf hi = *(const v8bf*)(p + 16);
  return __builtin_shufflevector(lo, hi, 0,1,2,3,4,5,6,7,8,9,10,11,12,13,14,15);
}

// B-matrix 32x16 bf16 fragment from B^T row-major [16 cols as rows x K].
// Lane L: col = L&15; halves 0..7 -> K = g*8.. ; halves 8..15 -> K = 16+g*8.. (g = L>>4)
template <typename P>
static DEV v16bf load_b(P base, int ld, int lane) {
  P p = base + (size_t)(lane & 15) * ld + (lane >> 4) * 8;
  v8bf lo = *(const v8bf*)(p);
  v8bf hi = *(const v8bf*)(p + 16);
  return __builtin_shufflevector(lo, hi, 0,1,2,3,4,5,6,7,8,9,10,11,12,13,14,15);
}

static DEV v8f wmma_bf16(v16bf a, v16bf b, v8f c) {
  return __builtin_amdgcn_wmma_f32_16x16x32_bf16(false, a, false, b, (short)0, c, false, false);
}

// ---- wave32 16-lane butterfly reductions via DPP16 ROW_XMASK (no LDS, no dscnt) ----
// mov_dpp has no tied 'old' operand -> no extra v_mov per stage.
template <int MASK>
static DEV float dpp_xor(float v) {
  int y = __builtin_amdgcn_mov_dpp(__builtin_bit_cast(int, v), 0x160 + MASK, 0xF, 0xF, false);
  return __builtin_bit_cast(float, y);
}
static DEV float rmax16(float v) {
  v = fmaxf(v, dpp_xor<1>(v));
  v = fmaxf(v, dpp_xor<2>(v));
  v = fmaxf(v, dpp_xor<4>(v));
  v = fmaxf(v, dpp_xor<8>(v));
  return v;
}
static DEV float rsum16(float v) {
  v += dpp_xor<1>(v);
  v += dpp_xor<2>(v);
  v += dpp_xor<4>(v);
  v += dpp_xor<8>(v);
  return v;
}

// ---- async global -> LDS (CDNA5), ASYNCcnt-tracked ----
// Generic pointers to __shared__ carry the LDS byte offset in bits [31:0] (ISA 10.2).
static DEV void async_b128(const bf16* lds_dst, const bf16* g_src) {
  unsigned l = (unsigned)(unsigned long long)lds_dst;
  unsigned long long g = (unsigned long long)g_src;
  asm volatile("global_load_async_to_lds_b128 %0, %1, off"
               :: "v"(l), "v"(g) : "memory");
}
static DEV void wait_async_le4() {
#if __has_builtin(__builtin_amdgcn_s_wait_asynccnt)
  __builtin_amdgcn_s_wait_asynccnt(4);
#else
  asm volatile("s_wait_asynccnt 4" ::: "memory");
#endif
}

// ---- kernel 1: f32 -> bf16 convert (8 elems/thread, b128 packed stores) ----
__global__ void cvt_f32_bf16(const float* __restrict__ in, bf16* __restrict__ out, int n) {
  int i = (blockIdx.x * blockDim.x + threadIdx.x) * 8;
  int stride = gridDim.x * blockDim.x * 8;
  for (; i < n; i += stride) {
    float4 a = *(const float4*)(in + i);
    float4 b = *(const float4*)(in + i + 4);
    v8bf o = { f2bf(a.x), f2bf(a.y), f2bf(a.z), f2bf(a.w),
               f2bf(b.x), f2bf(b.y), f2bf(b.z), f2bf(b.w) };
    *(v8bf*)(out + i) = o;
  }
}

// ---- kernels 2/3/5: C = A @ B^T (A [M,K] row-major bf16, Bw [N,K] row-major bf16) ----
// Block: 8 waves; each wave computes 32(M) x 64(N): 2 A-frags + 4 B-frags -> 8 WMMA/k-step.
enum { MODE_Q = 0, MODE_KV = 1, MODE_OUT = 2 };

template <int MODE>
__global__ void __launch_bounds__(256)
gemm_bt_kernel(const bf16* __restrict__ A, const bf16* __restrict__ Bw,
               bf16* __restrict__ out0, bf16* __restrict__ out1,
               float* __restrict__ outf, int M, int N, int K) {
  const int lane = threadIdx.x & 31;
  const int w    = threadIdx.x >> 5;
  const int m0   = blockIdx.x * 256 + w * 32;
  const int n0   = blockIdx.y * 64;

  v8f acc[2][4] = {};
  const bf16* Ap = A + (size_t)m0 * K;
  for (int k0 = 0; k0 < K; k0 += 32) {
    v16bf a0 = load_a(Ap + k0, K, lane);
    v16bf a1 = load_a(Ap + (size_t)16 * K + k0, K, lane);
#pragma unroll
    for (int nt = 0; nt < 4; ++nt) {
      v16bf bfv = load_b(Bw + (size_t)(n0 + nt * 16) * K + k0, K, lane);
      acc[0][nt] = wmma_bf16(a0, bfv, acc[0][nt]);
      acc[1][nt] = wmma_bf16(a1, bfv, acc[1][nt]);
    }
  }

  const int g = lane >> 4, nl = lane & 15;
  // In C layout: e = n0 + nt*16 + nl, so e&15 == nl (head) for all nt.
  if constexpr (MODE == MODE_Q) {
    const int d0 = n0 >> 4;            // e>>4 = d0 + nt : 4 consecutive d per lane
#pragma unroll
    for (int rt = 0; rt < 2; ++rt)
#pragma unroll
      for (int i = 0; i < 8; ++i) {
        int m = m0 + rt * 16 + i + 8 * g;
        int b = m >> 11, t = m & 2047;
        v4bf pk = { f2bf(acc[rt][0][i] * QSCALE), f2bf(acc[rt][1][i] * QSCALE),
                    f2bf(acc[rt][2][i] * QSCALE), f2bf(acc[rt][3][i] * QSCALE) };
        *(v4bf*)&out0[(((size_t)b * Hh + nl) * Sq + t) * Dh + d0] = pk;   // Qh[b,h,t,d0..d0+3]
      }
  } else if constexpr (MODE == MODE_KV) {
    const int d0 = n0 >> 5;            // nt 0/2 -> K at d0,d0+1 ; nt 1/3 -> V at d0,d0+1
#pragma unroll
    for (int rt = 0; rt < 2; ++rt)
#pragma unroll
      for (int i = 0; i < 8; ++i) {
        int m = m0 + rt * 16 + i + 8 * g;
        int b = m >> 11, t = m & 2047;
        v2bf pk = { f2bf(acc[rt][0][i]), f2bf(acc[rt][2][i]) };
        *(v2bf*)&out0[(((size_t)b * Hh + nl) * Sq + t) * Dh + d0] = pk;   // Kh[b,h,t,d0..d0+1]
        out1[(((size_t)b * Hh + nl) * Dh + d0) * Sq + t]     = f2bf(acc[rt][1][i]);  // V^T
        out1[(((size_t)b * Hh + nl) * Dh + d0 + 1) * Sq + t] = f2bf(acc[rt][3][i]);
      }
  } else {
#pragma unroll
    for (int rt = 0; rt < 2; ++rt)
#pragma unroll
      for (int nt = 0; nt < 4; ++nt)
#pragma unroll
        for (int i = 0; i < 8; ++i) {
          int m = m0 + rt * 16 + i + 8 * g;
          int e = n0 + nt * 16 + nl;
          outf[(size_t)m * N + e] = acc[rt][nt][i];
        }
  }
}

// ---- kernel 4: flash attention per (b,h) ----
// Block = 8 waves, 128 query rows. K/V^T tiles (64x64) double-buffered in LDS via
// async global->LDS DMA; wave owns 16 query rows; online softmax with DPP reductions.
__global__ void __launch_bounds__(256)
attn_kernel(const bf16* __restrict__ Qh, const bf16* __restrict__ Kh,
            const bf16* __restrict__ VhT, bf16* __restrict__ Ao) {
  __shared__ __align__(32) bf16 ldsK[2][64 * 64];   // [key j][d]
  __shared__ __align__(32) bf16 ldsV[2][64 * 64];   // [d][key j]
  __shared__ __align__(32) bf16 ldsP[8 * 16 * 64];  // per-wave P relayout
  const int tid  = threadIdx.x;
  const int lane = tid & 31;
  const int w    = tid >> 5;
  const int b = blockIdx.z, h = blockIdx.y;
  const int qbase = blockIdx.x * 128 + w * 16;
  const size_t bh = (size_t)b * Hh + h;
  const bf16* Qp = Qh + (bh * Sq + qbase) * Dh;
  const bf16* Kp = Kh + bh * (size_t)Sq * Dh;
  const bf16* Vp = VhT + bh * (size_t)Dh * Sq;

  // cooperative tile loader: thread covers row = tid>>2, elements (tid&3)*16 .. +15
  const int lrow = tid >> 2;
  const int lcol = (tid & 3) * 16;
  auto issue_tile = [&](int j, int bufi) {
    const bf16* gk = Kp + (size_t)(j + lrow) * Dh + lcol;
    const bf16* lk = &ldsK[bufi][lrow * 64 + lcol];
    async_b128(lk, gk);
    async_b128(lk + 8, gk + 8);
    const bf16* gv = Vp + (size_t)lrow * Sq + j + lcol;
    const bf16* lv = &ldsV[bufi][lrow * 64 + lcol];
    async_b128(lv, gv);
    async_b128(lv + 8, gv + 8);
  };

  v16bf qf0 = load_a(Qp, Dh, lane);        // Q already scaled by SCALE*log2(e)
  v16bf qf1 = load_a(Qp + 32, Dh, lane);

  v8f O[4] = {};
  float rmax[8], rsum[8];
#pragma unroll
  for (int i = 0; i < 8; ++i) { rmax[i] = -1e30f; rsum[i] = 0.f; }

  bf16* myP = ldsP + w * (16 * 64);
  const int g = lane >> 4, nl = lane & 15;

  issue_tile(0, 0);
  int buf = 0;
  for (int j0 = 0; j0 < Sq; j0 += 64) {
    int jn = (j0 + 64 < Sq) ? j0 + 64 : 0;   // last iter: dummy refill keeps wait imm uniform
    issue_tile(jn, buf ^ 1);
    wait_async_le4();                         // only the 4 just-issued may remain in flight
    __syncthreads();

    const bf16* Kt = ldsK[buf];
    const bf16* Vt = ldsV[buf];

    // S = Q K^T (logits already in log2 domain)
    v8f s[4] = {};
#pragma unroll
    for (int st = 0; st < 4; ++st) {
      v16bf kf0 = load_b(Kt + st * 16 * 64, 64, lane);
      v16bf kf1 = load_b(Kt + st * 16 * 64 + 32, 64, lane);
      s[st] = wmma_bf16(qf0, kf0, s[st]);
      s[st] = wmma_bf16(qf1, kf1, s[st]);
    }

    // online softmax (rows live in 16-lane groups; reduce via DPP ROW_XMASK)
#pragma unroll
    for (int i = 0; i < 8; ++i) {
      float bm = fmaxf(fmaxf(s[0][i], s[1][i]), fmaxf(s[2][i], s[3][i]));
      bm = rmax16(bm);
      float nm = fmaxf(rmax[i], bm);
      float sc = exp2f(rmax[i] - nm);
      rmax[i] = nm;
      float part = 0.f;
#pragma unroll
      for (int st = 0; st < 4; ++st) {
        float p = exp2f(s[st][i] - nm);
        s[st][i] = p;
        part += p;
      }
      rsum[i] = rsum[i] * sc + rsum16(part);
#pragma unroll
      for (int dt = 0; dt < 4; ++dt) O[dt][i] *= sc;
    }

    // relayout P (C-layout) -> LDS bf16 [16 rows x 64 keys] -> A fragments
#pragma unroll
    for (int st = 0; st < 4; ++st)
#pragma unroll
      for (int i = 0; i < 8; ++i)
        myP[(i + 8 * g) * 64 + st * 16 + nl] = f2bf(s[st][i]);
    v16bf pf0 = load_a(myP, 64, lane);
    v16bf pf1 = load_a(myP + 32, 64, lane);

    // O += P V  (V^T tile rows contiguous over keys -> B operand)
#pragma unroll
    for (int dt = 0; dt < 4; ++dt) {
      v16bf vf0 = load_b(Vt + dt * 16 * 64, 64, lane);
      v16bf vf1 = load_b(Vt + dt * 16 * 64 + 32, 64, lane);
      O[dt] = wmma_bf16(pf0, vf0, O[dt]);
      O[dt] = wmma_bf16(pf1, vf1, O[dt]);
    }
    __syncthreads();                          // protect buf^1 before next issue
    buf ^= 1;
  }

  // normalize and store in 'b t (h d)' layout
#pragma unroll
  for (int i = 0; i < 8; ++i) {
    float inv = 1.0f / rsum[i];
    int t = qbase + i + 8 * g;
#pragma unroll
    for (int dt = 0; dt < 4; ++dt) {
      int e = h * Dh + dt * 16 + nl;
      Ao[((size_t)b * Sq + t) * Ee + e] = f2bf(O[dt][i] * inv);
    }
  }
}

extern "C" void kernel_launch(void* const* d_in, const int* in_sizes, int n_in,
                              void* d_out, int out_size, void* d_ws, size_t ws_size,
                              hipStream_t stream) {
  (void)in_sizes; (void)n_in; (void)out_size; (void)ws_size;
  const float* q   = (const float*)d_in[0];
  const float* kv  = (const float*)d_in[1];
  const float* Wq  = (const float*)d_in[2];
  const float* Wkv = (const float*)d_in[3];
  const float* W0  = (const float*)d_in[4];
  float* out = (float*)d_out;

  char* ws = (char*)d_ws;
  size_t off = 0;
  auto alloc = [&](size_t bytes) -> void* {
    void* p = ws + off;
    off = (off + bytes + 255) & ~(size_t)255;
    return p;
  };
  const size_t NA = (size_t)Bt * Sq * Dm;
  bf16* qbf   = (bf16*)alloc(NA * 2);
  bf16* kvbf  = (bf16*)alloc(NA * 2);
  bf16* wqbf  = (bf16*)alloc((size_t)Dm * Dm * 2);
  bf16* wkvbf = (bf16*)alloc((size_t)2 * Dm * Dm * 2);
  bf16* w0bf  = (bf16*)alloc((size_t)Dm * Dm * 2);
  bf16* Qh    = (bf16*)alloc(NA * 2);
  bf16* Kh    = (bf16*)alloc(NA * 2);
  bf16* VhT   = (bf16*)alloc(NA * 2);
  bf16* Ao    = (bf16*)alloc(NA * 2);

  cvt_f32_bf16<<<1024, 256, 0, stream>>>(q,   qbf,   (int)NA);
  cvt_f32_bf16<<<1024, 256, 0, stream>>>(kv,  kvbf,  (int)NA);
  cvt_f32_bf16<<<512,  256, 0, stream>>>(Wq,  wqbf,  Dm * Dm);
  cvt_f32_bf16<<<512,  256, 0, stream>>>(Wkv, wkvbf, 2 * Dm * Dm);
  cvt_f32_bf16<<<512,  256, 0, stream>>>(W0,  w0bf,  Dm * Dm);

  const int M = Bt * Sq;  // 8192
  gemm_bt_kernel<MODE_Q><<<dim3(M / 256, Dm / 64), 256, 0, stream>>>(
      qbf, wqbf, Qh, nullptr, nullptr, M, Dm, Dm);
  gemm_bt_kernel<MODE_KV><<<dim3(M / 256, (2 * Dm) / 64), 256, 0, stream>>>(
      kvbf, wkvbf, Kh, VhT, nullptr, M, 2 * Dm, Dm);
  attn_kernel<<<dim3(Sq / 128, Hh, Bt), 256, 0, stream>>>(Qh, Kh, VhT, Ao);
  gemm_bt_kernel<MODE_OUT><<<dim3(M / 256, Dm / 64), 256, 0, stream>>>(
      Ao, w0bf, nullptr, nullptr, out, M, Dm, Dm);
}
